// MH_MambaBlock_82282983456879
// MI455X (gfx1250) — compile-verified, matrix operand order checked by reading, not run on hardware
//
#include <hip/hip_runtime.h>
#include <hip/hip_bf16.h>

typedef __attribute__((ext_vector_type(16))) __bf16 v16bf;
typedef __attribute__((ext_vector_type(8)))  float  v8f;

#define B_   16
#define DIM_ 128
#define H_   128
#define W_   128
#define NH_  4
#define HD_  32
#define HID_ 32
#define K_   5
#define TB_  16      // spatial tile 16x16
#define TP_  20      // tile + 2*halo
#define NPIXH_ 400   // 20*20
#define NPIX_  256   // 16*16

// ---- dynamic LDS layout (bytes) ----
#define OFF_WIN    0        // W_in  bf16 32x32          = 2048
#define OFF_WGV    2048     // W_gv  bf16 4x32x32        = 8192
#define OFF_WGH    10240    // W_gh  bf16 4x32x32        = 8192
#define OFF_WOUT   18432    // W_out bf16 128x128        = 32768
#define OFF_FCONST 51200    // f32 consts (1952 floats)  = 7808
#define OFF_CSUM   59008    // f32 256 (sum/sumsq)       = 1024
#define OFF_XBUF   60032    // bf16: x halo 400x128 (102400); reused: v 256x128 (65536) + h (65536)
#define OFF_XP     191104   // bf16: xp halo 400x128 (102400); reused: out_pre 256x128 (65536)
#define SMEM_BYTES 293504

// f32 const region sub-offsets (in floats)
#define FC_WV   0      // 640
#define FC_WH   640    // 640
#define FC_BV   1280   // 128
#define FC_BH   1408   // 128
#define FC_BIN  1536   // 32
#define FC_BGV  1568   // 128
#define FC_BGH  1696   // 128
#define FC_BOUT 1824   // 128

// Load a 16x32 bf16 WMMA fragment (A: M rows / B: N cols selected by lane%16,
// K runs contiguously: elements [k..k+7] and [k+16..k+23], hi-half lanes +8).
__device__ __forceinline__ v16bf ldfrag(const __bf16* base, int stride,
                                        int r_base, int k_base, int lane) {
  const int l16 = lane & 15;
  const int hioff = (lane & 16) ? 8 : 0;
  const __bf16* p = base + (r_base + l16) * stride + k_base + hioff;
  v16bf r;
#pragma unroll
  for (int i = 0; i < 8; ++i) { r[i] = p[i]; r[i + 8] = p[i + 16]; }
  return r;
}

__device__ __forceinline__ float sigmoid_f(float x) {
  return 1.0f / (1.0f + __expf(-x));
}

__global__ void __launch_bounds__(256)
mamba_zero_stats(float* gsum) {
  int t = threadIdx.x;
  if (t < 256) gsum[t] = 0.0f;
}

__global__ void __launch_bounds__(256)
mamba_fused_main(const float* __restrict__ x,
                 const float* __restrict__ W_in,  const float* __restrict__ b_in,
                 const float* __restrict__ W_v,   const float* __restrict__ b_v,
                 const float* __restrict__ W_h,   const float* __restrict__ b_h,
                 const float* __restrict__ W_gv,  const float* __restrict__ b_gv,
                 const float* __restrict__ W_gh,  const float* __restrict__ b_gh,
                 const float* __restrict__ W_out, const float* __restrict__ b_out,
                 float* __restrict__ y_out, float* __restrict__ gsum) {
  extern __shared__ char smem[];
  __bf16* w_in_s  = (__bf16*)(smem + OFF_WIN);
  __bf16* w_gv_s  = (__bf16*)(smem + OFF_WGV);
  __bf16* w_gh_s  = (__bf16*)(smem + OFF_WGH);
  __bf16* w_out_s = (__bf16*)(smem + OFF_WOUT);
  float*  fconst  = (float*)(smem + OFF_FCONST);
  float*  csum    = (float*)(smem + OFF_CSUM);
  __bf16* xbuf    = (__bf16*)(smem + OFF_XBUF);           // 400x128 halo x
  __bf16* vbuf    = (__bf16*)(smem + OFF_XBUF);           // reuse: 256x128 v
  __bf16* hbuf    = (__bf16*)(smem + OFF_XBUF + 65536);   // reuse: 256x128 h
  __bf16* xpbuf   = (__bf16*)(smem + OFF_XP);             // 400x128 halo xp
  __bf16* opre    = (__bf16*)(smem + OFF_XP);             // reuse: 256x128 gated

  const int tid  = threadIdx.x;
  const int lane = tid & 31;
  const int wave = tid >> 5;
  const int l16  = lane & 15;
  const int hi8  = (lane & 16) ? 8 : 0;
  const int x0 = blockIdx.x * TB_;
  const int y0 = blockIdx.y * TB_;
  const int bb = blockIdx.z;

  // ---------- phase 0: stage weights into LDS ----------
  for (int i = tid; i < 1024;  i += 256) w_in_s[i]  = (__bf16)W_in[i];
  for (int i = tid; i < 4096;  i += 256) w_gv_s[i]  = (__bf16)W_gv[i];
  for (int i = tid; i < 4096;  i += 256) w_gh_s[i]  = (__bf16)W_gh[i];
  for (int i = tid; i < 16384; i += 256) w_out_s[i] = (__bf16)W_out[i];
  for (int i = tid; i < 640; i += 256) { fconst[FC_WV + i] = W_v[i]; fconst[FC_WH + i] = W_h[i]; }
  for (int i = tid; i < 128; i += 256) {
    fconst[FC_BV + i]   = b_v[i];  fconst[FC_BH + i]   = b_h[i];
    fconst[FC_BGV + i]  = b_gv[i]; fconst[FC_BGH + i]  = b_gh[i];
    fconst[FC_BOUT + i] = b_out[i];
  }
  if (tid < 32)  fconst[FC_BIN + tid] = b_in[tid];
  if (tid < 256) csum[tid] = 0.0f;

  // ---------- phase 1: stage x halo tile (bf16, [pix][ch]) ----------
  for (int i = tid; i < NPIXH_ * DIM_; i += 256) {
    int ch = i / NPIXH_, pix = i % NPIXH_;
    int py = pix / TP_, px = pix % TP_;
    int gy = y0 - 2 + py, gx = x0 - 2 + px;
    float v = 0.0f;
    if (gy >= 0 && gy < H_ && gx >= 0 && gx < W_)
      v = x[(((size_t)bb * DIM_ + ch) * H_ + gy) * W_ + gx];
    xbuf[pix * DIM_ + ch] = (__bf16)v;
  }
  __syncthreads();

  // ---------- phase 2: in_proj via WMMA over all 400 halo pixels ----------
  // 25 N-tiles x 8 M-tiles (4 heads x 2); K = 32 = one wmma.
  for (int t = wave; t < 200; t += 8) {
    int nt = t % 25, mt = t / 25;
    int head = mt >> 1, mi = mt & 1;
    v16bf a = ldfrag(w_in_s, 32, mi * 16, 0, lane);
    v16bf b = ldfrag(xbuf, DIM_, nt * 16, head * HD_, lane);
    v8f c = {};
    c = __builtin_amdgcn_wmma_f32_16x16x32_bf16(false, a, false, b,
                                                (short)0, c, false, false);
    int pix = nt * 16 + l16;
#pragma unroll
    for (int r = 0; r < 8; ++r) {
      int cl = mi * 16 + r + hi8;
      xpbuf[pix * DIM_ + head * HID_ + cl] = (__bf16)(c[r] + fconst[FC_BIN + cl]);
    }
  }
  __syncthreads();

  // ---------- phase 3: depthwise K=5 vertical + horizontal ----------
  for (int e = tid; e < NPIX_ * DIM_; e += 256) {
    int pix = e >> 7, ch = e & 127;
    int py = pix >> 4, px = pix & 15;
    int head = ch >> 5, cl = ch & 31;
    int hy = py + 2, hx = px + 2;
    float sv = fconst[FC_BV + ch];
    float sh = fconst[FC_BH + ch];
#pragma unroll
    for (int j = 0; j < K_; ++j) {
      int gy = y0 + py + j - 2;
      if (gy >= 0 && gy < H_)
        sv += fconst[FC_WV + (head * HID_ + cl) * K_ + j] *
              (float)xpbuf[((hy + j - 2) * TP_ + hx) * DIM_ + ch];
      int gx = x0 + px + j - 2;
      if (gx >= 0 && gx < W_)
        sh += fconst[FC_WH + (head * HID_ + cl) * K_ + j] *
              (float)xpbuf[(hy * TP_ + hx + j - 2) * DIM_ + ch];
    }
    vbuf[pix * DIM_ + ch] = (__bf16)sv;   // overwrites x stage (no longer needed)
    hbuf[pix * DIM_ + ch] = (__bf16)sh;
  }
  __syncthreads();

  // ---------- phase 4: cross-gating (two WMMAs per tile) + combine ----------
  // 128 jobs: head(4) x mi(2) x nt(16)
  for (int t = wave; t < 128; t += 8) {
    int head = t >> 5, rem = t & 31;
    int mi = rem >> 4, nt = rem & 15;
    v16bf agv = ldfrag(w_gv_s + head * 1024, 32, mi * 16, 0, lane);
    v16bf bh  = ldfrag(hbuf, DIM_, nt * 16, head * HID_, lane);
    v8f cgv = {};
    cgv = __builtin_amdgcn_wmma_f32_16x16x32_bf16(false, agv, false, bh,
                                                  (short)0, cgv, false, false);
    v16bf agh = ldfrag(w_gh_s + head * 1024, 32, mi * 16, 0, lane);
    v16bf bv  = ldfrag(vbuf, DIM_, nt * 16, head * HID_, lane);
    v8f cgh = {};
    cgh = __builtin_amdgcn_wmma_f32_16x16x32_bf16(false, agh, false, bv,
                                                  (short)0, cgh, false, false);
    int pix = nt * 16 + l16;
#pragma unroll
    for (int r = 0; r < 8; ++r) {
      int cl = mi * 16 + r + hi8;
      int ch = head * HID_ + cl;
      float vv = (float)vbuf[pix * DIM_ + ch];
      float hh = (float)hbuf[pix * DIM_ + ch];
      float g1 = sigmoid_f(cgv[r] + fconst[FC_BGV + ch]);
      float g2 = sigmoid_f(cgh[r] + fconst[FC_BGH + ch]);
      opre[pix * DIM_ + ch] = (__bf16)(vv * g1 + hh * g2);
    }
  }
  __syncthreads();

  // ---------- phase 5: out_proj (K=128 -> 4 WMMAs) + residual + BN stats ----------
  for (int t = wave; t < 128; t += 8) {
    int mt = t >> 4, nt = t & 15;
    int m_base = mt * 16;
    v8f c = {};
#pragma unroll
    for (int kb = 0; kb < DIM_; kb += 32) {
      v16bf a = ldfrag(w_out_s, DIM_, m_base, kb, lane);
      v16bf b = ldfrag(opre, DIM_, nt * 16, kb, lane);
      c = __builtin_amdgcn_wmma_f32_16x16x32_bf16(false, a, false, b,
                                                  (short)0, c, false, false);
    }
    int pix = nt * 16 + l16;
    int py = pix >> 4, px = pix & 15;
    int gy = y0 + py, gx = x0 + px;
#pragma unroll
    for (int r = 0; r < 8; ++r) {
      int ch = m_base + r + hi8;
      size_t gidx = (((size_t)bb * DIM_ + ch) * H_ + gy) * W_ + gx;
      float yv = c[r] + fconst[FC_BOUT + ch] + x[gidx];
      y_out[gidx] = yv;
      atomicAdd(&csum[ch], yv);
      atomicAdd(&csum[128 + ch], yv * yv);
    }
  }
  __syncthreads();

  // per-block -> global partial reduction (two f32 atomics per channel per block: sum, sumsq)
  if (tid < 256) atomicAdd(&gsum[tid], csum[tid]);
}

__global__ void __launch_bounds__(128)
mamba_bn_finalize(const float* __restrict__ gsum,
                  const float* __restrict__ gamma,
                  const float* __restrict__ beta,
                  float* __restrict__ scale_shift) {
  int c = threadIdx.x;
  if (c < DIM_) {
    const float invN = 1.0f / (float)(B_ * H_ * W_);
    float mean = gsum[c] * invN;
    float var  = gsum[128 + c] * invN - mean * mean;
    float sc = gamma[c] * rsqrtf(var + 1e-5f);
    scale_shift[c]        = sc;
    scale_shift[128 + c]  = beta[c] - mean * sc;
  }
}

__global__ void __launch_bounds__(256)
mamba_bn_apply(float* __restrict__ y, const float* __restrict__ scale_shift) {
  const size_t n4 = (size_t)B_ * DIM_ * H_ * W_ / 4;
  float4* y4 = (float4*)y;
  for (size_t i = blockIdx.x * blockDim.x + threadIdx.x; i < n4;
       i += (size_t)gridDim.x * blockDim.x) {
    int ch = (int)((i * 4) >> 14) & 127;   // H*W = 16384, same ch across float4
    float sc = scale_shift[ch], sh = scale_shift[128 + ch];
    float4 v = y4[i];
    v.x = v.x * sc + sh; v.y = v.y * sc + sh;
    v.z = v.z * sc + sh; v.w = v.w * sc + sh;
    y4[i] = v;
  }
}

extern "C" void kernel_launch(void* const* d_in, const int* in_sizes, int n_in,
                              void* d_out, int out_size, void* d_ws, size_t ws_size,
                              hipStream_t stream) {
  const float* x     = (const float*)d_in[0];
  const float* W_in  = (const float*)d_in[1];
  const float* b_in  = (const float*)d_in[2];
  const float* W_v   = (const float*)d_in[3];
  const float* b_v   = (const float*)d_in[4];
  const float* W_h   = (const float*)d_in[5];
  const float* b_h   = (const float*)d_in[6];
  const float* W_gv  = (const float*)d_in[7];
  const float* b_gv  = (const float*)d_in[8];
  const float* W_gh  = (const float*)d_in[9];
  const float* b_gh  = (const float*)d_in[10];
  const float* W_out = (const float*)d_in[11];
  const float* b_out = (const float*)d_in[12];
  const float* gamma = (const float*)d_in[13];
  const float* beta  = (const float*)d_in[14];
  float* y = (float*)d_out;

  float* gsum        = (float*)d_ws;        // [256]: sum + sumsq
  float* scale_shift = gsum + 256;          // [256]: scale + shift

  mamba_zero_stats<<<1, 256, 0, stream>>>(gsum);

  dim3 grid(W_ / TB_, H_ / TB_, B_);        // 8 x 8 x 16
  mamba_fused_main<<<grid, 256, SMEM_BYTES, stream>>>(
      x, W_in, b_in, W_v, b_v, W_h, b_h, W_gv, b_gv, W_gh, b_gh,
      W_out, b_out, y, gsum);

  mamba_bn_finalize<<<1, 128, 0, stream>>>(gsum, gamma, beta, scale_shift);
  mamba_bn_apply<<<2048, 256, 0, stream>>>(y, scale_shift);
}
